// thickness_Autoencoder_67551245632215
// MI455X (gfx1250) — matrix operand order repaired and checked by reading, not compiled, not run on hardware
//
#include <hip/hip_runtime.h>
#include <stdint.h>
#include <stddef.h>

// ---------------------------------------------------------------------------
// CDNA5 WMMA types (wave32): A,B = 16 halves/lane (16x32 / 32x16 f16),
// C/D = 8 floats/lane (16x16 f32).
// ---------------------------------------------------------------------------
typedef __attribute__((ext_vector_type(16))) _Float16 v16h;
typedef __attribute__((ext_vector_type(8)))  _Float16 v8h;
typedef __attribute__((ext_vector_type(8)))  float    v8f;

// A-fragment (16x32 f16) K index for lane-half khalf, half-slot j.
// Lanes 0-15 hold K=0..15 ; lanes 16-31 hold K=16..31.
__device__ __forceinline__ int a_frag_k(int khalf, int j) {
  return khalf * 16 + j;
}

// ===========================================================================
// Weight pre-pack: conv weights w[CO][CI][3][3] (f32) -> A-fragment f16,
// K enumerated TAP-MAJOR: k = (ky*3+kx)*CI + ci  (matches NHWC B gather).
// dst layout: [(coTile*KSTEPS + ks)*32 + lane]*16 + j  (contig 32B per lane)
// ===========================================================================
__global__ void packA_conv_kernel(const float* __restrict__ w, _Float16* __restrict__ dst,
                                  int CO, int CI, int total) {
  int tid = blockIdx.x * blockDim.x + threadIdx.x;
  if (tid >= total) return;
  int lane = tid & 31;
  int t1   = tid >> 5;
  int KSTEPS = (CI * 9) / 32;
  int ks = t1 % KSTEPS;
  int ct = t1 / KSTEPS;
  int m = lane & 15, khalf = lane >> 4;
  int co = ct * 16 + m;
  #pragma unroll
  for (int j = 0; j < 16; ++j) {
    int kk = ks * 32 + a_frag_k(khalf, j);
    int kt = kk / CI;          // tap index 0..8  (ky*3+kx)
    int ci = kk % CI;
    float v = w[((size_t)co * CI + ci) * 9 + kt];
    dst[((size_t)(ct * KSTEPS + ks) * 32 + lane) * 16 + j] = (_Float16)v;
  }
}

// Transposed-conv weight pre-pack for one output-parity class (py,px).
// dw layout [CI][CO][3][3]; K index = tap*CI + ci (tap-major).
// ky: py==0 -> 1 ; py==1 -> (ty==0 ? 2 : 0). Same for kx.
__global__ void packA_dconv_kernel(const float* __restrict__ dw, _Float16* __restrict__ dst,
                                   int CI, int CO, int py, int px, int total) {
  int tid = blockIdx.x * blockDim.x + threadIdx.x;
  if (tid >= total) return;
  int lane = tid & 31;
  int t1   = tid >> 5;
  int ntx = 1 + px;
  int nt  = (1 + py) * ntx;
  int K = CI * nt;
  int KSTEPS = K / 32;
  int ks = t1 % KSTEPS;
  int ct = t1 / KSTEPS;
  int m = lane & 15, khalf = lane >> 4;
  int co = ct * 16 + m;
  #pragma unroll
  for (int j = 0; j < 16; ++j) {
    int kk = ks * 32 + a_frag_k(khalf, j);
    int t  = kk / CI;
    int ci = kk % CI;
    int tx = t & (ntx - 1);
    int ty = t >> (ntx - 1);
    int ky = (py == 0) ? 1 : ((ty == 0) ? 2 : 0);
    int kx = (px == 0) ? 1 : ((tx == 0) ? 2 : 0);
    float v = dw[(size_t)(ci * CO + co) * 9 + ky * 3 + kx];
    dst[((size_t)(ct * KSTEPS + ks) * 32 + lane) * 16 + j] = (_Float16)v;
  }
}

// Pack VQ flat rows into A-fragment order.  Flat row grow = (b*128+c)*16+g
// covers spatial positions g*64 .. g*64+63 at fixed channel c (NCHW reshape);
// source is NHWC f16 [b][32][32][128] -> strided gather (one-shot reorder).
__global__ void packA_rows_kernel(const _Float16* __restrict__ src, _Float16* __restrict__ dst,
                                  int total) {
  int tid = blockIdx.x * blockDim.x + threadIdx.x;
  if (tid >= total) return;
  int lane = tid & 31;
  int t1   = tid >> 5;
  int ks = t1 & 1;
  int rb = t1 >> 1;
  int m = lane & 15, khalf = lane >> 4;
  int grow = rb * 16 + m;
  int b = grow >> 11;          // 2048 rows per image
  int rc = grow & 2047;
  int c = rc >> 4;
  int g = rc & 15;
  #pragma unroll
  for (int j = 0; j < 16; ++j) {
    int kk = ks * 32 + a_frag_k(khalf, j);     // spatial offset within row
    int s = g * 64 + kk;
    dst[((size_t)(rb * 2 + ks) * 32 + lane) * 16 + j] =
        src[((size_t)b * 1024 + s) * 128 + c];
  }
}

__global__ void cvt_f32_to_f16_kernel(const float* __restrict__ src, _Float16* __restrict__ dst,
                                      int total) {
  int tid = blockIdx.x * blockDim.x + threadIdx.x;
  if (tid < total) dst[tid] = (_Float16)src[tid];
}

__global__ void emb_norms_kernel(const float* __restrict__ emb, float* __restrict__ norms) {
  int e = blockIdx.x * blockDim.x + threadIdx.x;
  if (e >= 512) return;
  float s = 0.0f;
  for (int k = 0; k < 64; ++k) { float v = emb[e * 64 + k]; s += v * v; }
  norms[e] = s;
}

// dw3 [32][1][3][3] -> tap-major f16: dst[(ky*3+kx)*32 + ci]
__global__ void pack_dw3_kernel(const float* __restrict__ dw, _Float16* __restrict__ dst) {
  int tid = blockIdx.x * blockDim.x + threadIdx.x;
  if (tid >= 9 * 32) return;
  int kt = tid >> 5, ci = tid & 31;
  dst[kt * 32 + ci] = (_Float16)dw[ci * 9 + kt];
}

// ===========================================================================
// conv1: 2->32, 3x3, stride2, pad1 (K=18 too small for WMMA).
// f32 NCHW in, f16 NHWC out [64][128][128][32]; channel-fastest threads.
// ===========================================================================
__global__ void conv1_kernel(const float* __restrict__ x, const float* __restrict__ w,
                             const float* __restrict__ bias, _Float16* __restrict__ outp,
                             int total) {
  int tid = blockIdx.x * blockDim.x + threadIdx.x;
  if (tid >= total) return;
  int co = tid & 31; int t = tid >> 5;
  int ox = t & 127;  t >>= 7;
  int oy = t & 127;  int bb = t >> 7;
  float s = bias[co];
  #pragma unroll
  for (int ci = 0; ci < 2; ++ci) {
    #pragma unroll
    for (int ky = 0; ky < 3; ++ky) {
      int iy = 2 * oy - 1 + ky;
      if ((unsigned)iy >= 256u) continue;
      #pragma unroll
      for (int kx = 0; kx < 3; ++kx) {
        int ix = 2 * ox - 1 + kx;
        if ((unsigned)ix >= 256u) continue;
        s += x[((size_t)(bb * 2 + ci) * 256 + iy) * 256 + ix] *
             w[((co * 2 + ci) * 3 + ky) * 3 + kx];
      }
    }
  }
  s = s > 0.0f ? s : 0.0f;
  outp[tid] = (_Float16)s;   // tid already == ((b*128+oy)*128+ox)*32+co
}

// ===========================================================================
// Implicit-GEMM conv 3x3 pad1 stride S on NHWC f16, v_wmma_f32_16x16x32_f16.
// One wave = TWO 16-channel co-tiles (ct, ct+CO/32) x 16 output pixels, so
// each B (activation) fragment is loaded once and used by 2 WMMAs.
// K tap-major: each lane's 16-wide B chunk = 16 consecutive channels at one
// tap -> single v16h (2x b128) load, bounds-checked once per tap.
// ===========================================================================
template <int CI, int S>
__global__ __launch_bounds__(128)
void conv_wmma_kernel(const _Float16* __restrict__ in, const _Float16* __restrict__ wpack,
                      const float* __restrict__ bias, _Float16* __restrict__ outp,
                      int B, int CO, int Hin, int Win, int Hout, int Wout, int totWaves) {
  int lane = threadIdx.x & 31;
  int wid  = (blockIdx.x * blockDim.x + threadIdx.x) >> 5;
  if (wid >= totWaves) return;

  int pixTiles = (Hout * Wout) / 16;
  int coPairs  = CO / 32;
  int cp = wid % coPairs;
  int t1 = wid / coPairs;
  int pt = t1 % pixTiles;
  int bb = t1 / pixTiles;
  int ct0 = cp, ct1 = cp + coPairs;           // two co-tiles per wave

  int n = lane & 15, khalf = lane >> 4;
  int pix = pt * 16 + n;
  int oy = pix / Wout, ox = pix % Wout;

  constexpr int K = CI * 9;
  constexpr int KSTEPS = K / 32;
  constexpr int CLOG = (CI == 128) ? 7 : ((CI == 64) ? 6 : 5);
  const _Float16* wp0 = wpack + (size_t)ct0 * KSTEPS * 512 + lane * 16;
  const _Float16* wp1 = wpack + (size_t)ct1 * KSTEPS * 512 + lane * 16;

  v8f acc0 = {}, acc1 = {};
  for (int ks = 0; ks < KSTEPS; ++ks) {
    v16h a0 = *(const v16h*)(wp0 + (size_t)ks * 512);
    v16h a1 = *(const v16h*)(wp1 + (size_t)ks * 512);
    int kbase = ks * 32 + khalf * 16;          // B frag: 16 consecutive K per lane
    int kt  = kbase >> CLOG;                   // tap (chunk never crosses a tap)
    int ci0 = kbase & (CI - 1);
    int ky = kt / 3, kx = kt % 3;
    int iy = oy * S - 1 + ky;
    int ix = ox * S - 1 + kx;
    v16h bf = {};
    if ((unsigned)iy < (unsigned)Hin && (unsigned)ix < (unsigned)Win)
      bf = *(const v16h*)(in + ((size_t)(bb * Hin + iy) * Win + ix) * CI + ci0);
    acc0 = __builtin_amdgcn_wmma_f32_16x16x32_f16(false, a0, false, bf, (short)0, acc0,
                                                  false, false);
    acc1 = __builtin_amdgcn_wmma_f32_16x16x32_f16(false, a1, false, bf, (short)0, acc1,
                                                  false, false);
  }
  // Epilogue: bias + ReLU -> one 16B NHWC store per co-tile (8 ch contiguous).
  size_t pbase = ((size_t)(bb * Hout + oy) * Wout + ox) * CO;
  v8h o0, o1;
  #pragma unroll
  for (int r = 0; r < 8; ++r) {
    float v0 = acc0[r] + bias[ct0 * 16 + khalf * 8 + r];
    float v1 = acc1[r] + bias[ct1 * 16 + khalf * 8 + r];
    o0[r] = (_Float16)(v0 > 0.0f ? v0 : 0.0f);
    o1[r] = (_Float16)(v1 > 0.0f ? v1 : 0.0f);
  }
  *(v8h*)(outp + pbase + ct0 * 16 + khalf * 8) = o0;
  *(v8h*)(outp + pbase + ct1 * 16 + khalf * 8) = o1;
}

// ===========================================================================
// Transposed conv 3x3 s2 p1 op1 on NHWC f16, one output-parity class/launch,
// two co-tiles per wave (same register blocking as conv_wmma).
// out[2y+py][2x+px] = sum_{(ty,tx) in [0,py]x[0,px], ci}
//     in[y+ty][x+tx][ci] * dw[ci][co][ky(ty)][kx(tx)];  K = CI*nTaps.
// ===========================================================================
template <int CI>
__global__ __launch_bounds__(128)
void dconv_wmma_kernel(const _Float16* __restrict__ in, const _Float16* __restrict__ wpack,
                       const float* __restrict__ bias, _Float16* __restrict__ outp,
                       int B, int CO, int Hin, int Win, int py, int px, int totWaves) {
  int lane = threadIdx.x & 31;
  int wid  = (blockIdx.x * blockDim.x + threadIdx.x) >> 5;
  if (wid >= totWaves) return;

  int ntx = 1 + px;
  int KSTEPS = (CI * (1 + py) * ntx) / 32;
  int pixTiles = (Hin * Win) / 16;
  int coPairs  = CO / 32;
  int cp = wid % coPairs;
  int t1 = wid / coPairs;
  int pt = t1 % pixTiles;
  int bb = t1 / pixTiles;
  int ct0 = cp, ct1 = cp + coPairs;

  int n = lane & 15, khalf = lane >> 4;
  int pix = pt * 16 + n;
  int y = pix / Win, x = pix % Win;
  int Hout = Hin * 2, Wout = Win * 2;
  constexpr int CLOG = (CI == 128) ? 7 : 6;

  const _Float16* wp0 = wpack + (size_t)ct0 * KSTEPS * 512 + lane * 16;
  const _Float16* wp1 = wpack + (size_t)ct1 * KSTEPS * 512 + lane * 16;

  v8f acc0 = {}, acc1 = {};
  for (int ks = 0; ks < KSTEPS; ++ks) {
    v16h a0 = *(const v16h*)(wp0 + (size_t)ks * 512);
    v16h a1 = *(const v16h*)(wp1 + (size_t)ks * 512);
    int kbase = ks * 32 + khalf * 16;
    int t   = kbase >> CLOG;                   // tap
    int ci0 = kbase & (CI - 1);
    int tx = t & (ntx - 1);
    int ty = t >> (ntx - 1);
    int iy = y + ty, ix = x + tx;
    v16h bf = {};
    if (iy < Hin && ix < Win)
      bf = *(const v16h*)(in + ((size_t)(bb * Hin + iy) * Win + ix) * CI + ci0);
    acc0 = __builtin_amdgcn_wmma_f32_16x16x32_f16(false, a0, false, bf, (short)0, acc0,
                                                  false, false);
    acc1 = __builtin_amdgcn_wmma_f32_16x16x32_f16(false, a1, false, bf, (short)0, acc1,
                                                  false, false);
  }
  int oy = 2 * y + py, ox = 2 * x + px;
  size_t pbase = ((size_t)(bb * Hout + oy) * Wout + ox) * CO;
  v8h o0, o1;
  #pragma unroll
  for (int r = 0; r < 8; ++r) {
    float v0 = acc0[r] + bias[ct0 * 16 + khalf * 8 + r];
    float v1 = acc1[r] + bias[ct1 * 16 + khalf * 8 + r];
    o0[r] = (_Float16)(v0 > 0.0f ? v0 : 0.0f);
    o1[r] = (_Float16)(v1 > 0.0f ? v1 : 0.0f);
  }
  *(v8h*)(outp + pbase + ct0 * 16 + khalf * 8) = o0;
  *(v8h*)(outp + pbase + ct1 * 16 + khalf * 8) = o1;
}

// ===========================================================================
// VQ: per flat row (64-dim), argmin over 512 codes of (||e||^2 - 2 f.e),
// then scatter codebook row back into the NHWC feature map.
// One wave = 16 rows x all 512 codes (32 WMMA tiles x 2 K-steps).
// ===========================================================================
__global__ __launch_bounds__(128)
void vq_kernel(const _Float16* __restrict__ encP, const _Float16* __restrict__ embh,
               const float* __restrict__ enorm, _Float16* __restrict__ outq, int totWaves) {
  int lane = threadIdx.x & 31;
  int wid  = (blockIdx.x * blockDim.x + threadIdx.x) >> 5;
  if (wid >= totWaves) return;
  int rb = wid;
  int n = lane & 15, khalf = lane >> 4;

  v16h a0 = *(const v16h*)(encP + ((size_t)(rb * 2 + 0) * 32 + lane) * 16);
  v16h a1 = *(const v16h*)(encP + ((size_t)(rb * 2 + 1) * 32 + lane) * 16);

  float bestv[8];
  int   besti[8];
  #pragma unroll
  for (int r = 0; r < 8; ++r) { bestv[r] = 3.4e38f; besti[r] = 0; }

  for (int et = 0; et < 32; ++et) {
    int code = et * 16 + n;
    v16h e0 = *(const v16h*)(embh + (size_t)code * 64 + khalf * 16);
    v16h e1 = *(const v16h*)(embh + (size_t)code * 64 + 32 + khalf * 16);
    v8f acc = {};
    acc = __builtin_amdgcn_wmma_f32_16x16x32_f16(false, a0, false, e0, (short)0, acc,
                                                 false, false);
    acc = __builtin_amdgcn_wmma_f32_16x16x32_f16(false, a1, false, e1, (short)0, acc,
                                                 false, false);
    float nm = enorm[code];
    #pragma unroll
    for (int r = 0; r < 8; ++r) {
      float s = nm - 2.0f * acc[r];
      if (s < bestv[r]) { bestv[r] = s; besti[r] = code; }
    }
  }

  // Butterfly min-reduce across the 16 lanes holding each row (wave32).
  #pragma unroll
  for (int m = 1; m < 16; m <<= 1) {
    #pragma unroll
    for (int r = 0; r < 8; ++r) {
      float ov = __shfl_xor(bestv[r], m, 32);
      int   oi = __shfl_xor(besti[r], m, 32);
      if (ov < bestv[r] || (ov == bestv[r] && oi < besti[r])) {
        bestv[r] = ov; besti[r] = oi;
      }
    }
  }

  // Rows 0-7 answer in lanes 0-15, rows 8-15 in lanes 16-31.  Scatter the
  // chosen codebook rows into NHWC: row grow=(b*128+c)*16+g covers spatial
  // g*64+col at channel c ->  out[((b*1024 + g*64 + col))*128 + c].
  #pragma unroll
  for (int r = 0; r < 8; ++r) {
    int iA = __shfl(besti[r], 0, 32);    // row r
    int iB = __shfl(besti[r], 16, 32);   // row r + 8
    int growA = rb * 16 + r;
    int growB = growA + 8;
    int bA = growA >> 11, cA = (growA & 2047) >> 4, gA = growA & 15;
    int bB = growB >> 11, cB = (growB & 2047) >> 4, gB = growB & 15;
    size_t baseA = ((size_t)bA * 1024 + gA * 64) * 128 + cA;
    size_t baseB = ((size_t)bB * 1024 + gB * 64) * 128 + cB;
    outq[baseA + (size_t)lane * 128]        = embh[(size_t)iA * 64 + lane];
    outq[baseA + (size_t)(lane + 32) * 128] = embh[(size_t)iA * 64 + 32 + lane];
    outq[baseB + (size_t)lane * 128]        = embh[(size_t)iB * 64 + lane];
    outq[baseB + (size_t)(lane + 32) * 128] = embh[(size_t)iB * 64 + 32 + lane];
  }
}

// ===========================================================================
// dconv3: 32->1 (Cout=1 too skinny for WMMA).  NHWC f16 in (32 contiguous
// channels -> v16h loads), tap-major packed f16 weights, f32 out.
// ===========================================================================
__global__ void dconv3_kernel(const _Float16* __restrict__ in, const _Float16* __restrict__ w3p,
                              const float* __restrict__ bias, float* __restrict__ outp,
                              int total) {
  int tid = blockIdx.x * blockDim.x + threadIdx.x;
  if (tid >= total) return;
  int ox = tid & 255; int t = tid >> 8;
  int oy = t & 255;   int bb = t >> 8;
  int py = oy & 1, px = ox & 1, y = oy >> 1, x = ox >> 1;
  float s = bias[0];
  for (int ty = 0; ty <= py; ++ty) {
    int iy = y + ty;
    if (iy >= 128) continue;
    int ky = (py == 0) ? 1 : ((ty == 0) ? 2 : 0);
    for (int tx = 0; tx <= px; ++tx) {
      int ix = x + tx;
      if (ix >= 128) continue;
      int kx = (px == 0) ? 1 : ((tx == 0) ? 2 : 0);
      const v16h* ip = (const v16h*)(in + ((size_t)(bb * 128 + iy) * 128 + ix) * 32);
      const v16h* wp = (const v16h*)(w3p + (ky * 3 + kx) * 32);
      v16h i0 = ip[0], i1 = ip[1];
      v16h w0 = wp[0], w1 = wp[1];
      float acc = 0.0f;
      #pragma unroll
      for (int e = 0; e < 16; ++e)
        acc += (float)i0[e] * (float)w0[e] + (float)i1[e] * (float)w1[e];
      s += acc;
    }
  }
  outp[((size_t)bb * 256 + oy) * 256 + ox] = s;
}

// ===========================================================================
// Host orchestration
// ===========================================================================
extern "C" void kernel_launch(void* const* d_in, const int* in_sizes, int n_in,
                              void* d_out, int out_size, void* d_ws, size_t ws_size,
                              hipStream_t stream) {
  (void)in_sizes; (void)n_in; (void)out_size; (void)ws_size;

  const float* x   = (const float*)d_in[0];
  const float* w1  = (const float*)d_in[1];
  const float* b1  = (const float*)d_in[2];
  const float* w2  = (const float*)d_in[3];
  const float* b2  = (const float*)d_in[4];
  const float* w3  = (const float*)d_in[5];
  const float* b3  = (const float*)d_in[6];
  const float* w4  = (const float*)d_in[7];
  const float* b4  = (const float*)d_in[8];
  const float* emb = (const float*)d_in[9];
  const float* dw1 = (const float*)d_in[10];
  const float* db1 = (const float*)d_in[11];
  const float* dw2 = (const float*)d_in[12];
  const float* db2 = (const float*)d_in[13];
  const float* dw3 = (const float*)d_in[14];
  const float* db3 = (const float*)d_in[15];
  float* out = (float*)d_out;

  const int NB = 64;  // batch
  char* ws = (char*)d_ws;
  size_t off = 0;
  auto carve = [&](size_t bytes) -> size_t {
    size_t o = off;
    off += (bytes + 255) & ~(size_t)255;
    return o;
  };

  size_t o_h1   = carve((size_t)NB * 32 * 128 * 128 * 2);  // conv1 out, NHWC f16
  size_t o_h2   = carve((size_t)NB * 64 * 64 * 64 * 2);    // conv2 out
  size_t o_h3   = carve((size_t)NB * 128 * 32 * 32 * 2);   // conv3 out
  size_t o_enc  = carve((size_t)NB * 128 * 32 * 32 * 2);   // conv4 out (encoded)
  size_t o_encP = carve((size_t)NB * 128 * 32 * 32 * 2);   // packed VQ A rows
  size_t o_vq   = carve((size_t)NB * 128 * 32 * 32 * 2);   // quantized (NHWC)
  size_t o_wp2  = carve((size_t)64 * 288 * 2);
  size_t o_wp3  = carve((size_t)128 * 576 * 2);
  size_t o_wp4  = carve((size_t)128 * 1152 * 2);
  size_t o_dp1[4], o_dp2[4];
  for (int c = 0; c < 4; ++c) {
    int nt = (1 + (c >> 1)) * (1 + (c & 1));
    o_dp1[c] = carve((size_t)64 * 128 * nt * 2);
    o_dp2[c] = carve((size_t)32 * 64 * nt * 2);
  }
  size_t o_embh = carve((size_t)512 * 64 * 2);
  size_t o_norm = carve((size_t)512 * 4);
  size_t o_w3p  = carve((size_t)9 * 32 * 2);
  size_t o_d1 = o_h2;  // reuse: h2 dead after conv3
  size_t o_d2 = o_h1;  // reuse: h1 dead after conv2

  _Float16* h1   = (_Float16*)(ws + o_h1);
  _Float16* h2   = (_Float16*)(ws + o_h2);
  _Float16* h3   = (_Float16*)(ws + o_h3);
  _Float16* enc  = (_Float16*)(ws + o_enc);
  _Float16* encP = (_Float16*)(ws + o_encP);
  _Float16* vqf  = (_Float16*)(ws + o_vq);
  _Float16* wp2  = (_Float16*)(ws + o_wp2);
  _Float16* wp3  = (_Float16*)(ws + o_wp3);
  _Float16* wp4  = (_Float16*)(ws + o_wp4);
  _Float16* embh = (_Float16*)(ws + o_embh);
  float*    enrm = (float*)(ws + o_norm);
  _Float16* w3p  = (_Float16*)(ws + o_w3p);
  _Float16* d1   = (_Float16*)(ws + o_d1);
  _Float16* d2   = (_Float16*)(ws + o_d2);

  // ---- Encoder ----
  {
    int total = NB * 32 * 128 * 128;
    conv1_kernel<<<total / 256, 256, 0, stream>>>(x, w1, b1, h1, total);
  }
  {
    int total = (64 / 16) * (288 / 32) * 32;
    packA_conv_kernel<<<(total + 255) / 256, 256, 0, stream>>>(w2, wp2, 64, 32, total);
    int waves = NB * ((64 * 64) / 16) * (64 / 32);   // 2 co-tiles per wave
    conv_wmma_kernel<32, 2><<<waves / 4, 128, 0, stream>>>(h1, wp2, b2, h2, NB, 64,
                                                           128, 128, 64, 64, waves);
  }
  {
    int total = (128 / 16) * (576 / 32) * 32;
    packA_conv_kernel<<<(total + 255) / 256, 256, 0, stream>>>(w3, wp3, 128, 64, total);
    int waves = NB * ((32 * 32) / 16) * (128 / 32);
    conv_wmma_kernel<64, 2><<<waves / 4, 128, 0, stream>>>(h2, wp3, b3, h3, NB, 128,
                                                           64, 64, 32, 32, waves);
  }
  {
    int total = (128 / 16) * (1152 / 32) * 32;
    packA_conv_kernel<<<(total + 255) / 256, 256, 0, stream>>>(w4, wp4, 128, 128, total);
    int waves = NB * ((32 * 32) / 16) * (128 / 32);
    conv_wmma_kernel<128, 1><<<waves / 4, 128, 0, stream>>>(h3, wp4, b4, enc, NB, 128,
                                                            32, 32, 32, 32, waves);
  }

  // ---- Vector quantizer ----
  cvt_f32_to_f16_kernel<<<(512 * 64) / 256, 256, 0, stream>>>(emb, embh, 512 * 64);
  emb_norms_kernel<<<2, 256, 0, stream>>>(emb, enrm);
  {
    int rowBlocks = (NB * 128 * 32 * 32 / 64) / 16;  // 8192 blocks of 16 rows
    int total = rowBlocks * 2 * 32;
    packA_rows_kernel<<<(total + 255) / 256, 256, 0, stream>>>(enc, encP, total);
    vq_kernel<<<rowBlocks / 4, 128, 0, stream>>>(encP, embh, enrm, vqf, rowBlocks);
  }

  // ---- Decoder ----
  for (int c = 0; c < 4; ++c) {  // dconv1: 128 -> 64, 32x32 -> 64x64
    int py = c >> 1, px = c & 1;
    int nt = (1 + py) * (1 + px);
    int total = (64 / 16) * ((128 * nt) / 32) * 32;
    packA_dconv_kernel<<<(total + 255) / 256, 256, 0, stream>>>(
        dw1, (_Float16*)(ws + o_dp1[c]), 128, 64, py, px, total);
    int waves = NB * ((32 * 32) / 16) * (64 / 32);
    dconv_wmma_kernel<128><<<waves / 4, 128, 0, stream>>>(
        vqf, (_Float16*)(ws + o_dp1[c]), db1, d1, NB, 64, 32, 32, py, px, waves);
  }
  for (int c = 0; c < 4; ++c) {  // dconv2: 64 -> 32, 64x64 -> 128x128
    int py = c >> 1, px = c & 1;
    int nt = (1 + py) * (1 + px);
    int total = (32 / 16) * ((64 * nt) / 32) * 32;
    packA_dconv_kernel<<<(total + 255) / 256, 256, 0, stream>>>(
        dw2, (_Float16*)(ws + o_dp2[c]), 64, 32, py, px, total);
    int waves = NB * ((64 * 64) / 16) * (32 / 32);
    dconv_wmma_kernel<64><<<waves / 4, 128, 0, stream>>>(
        d1, (_Float16*)(ws + o_dp2[c]), db2, d2, NB, 32, 64, 64, py, px, waves);
  }
  {
    pack_dw3_kernel<<<2, 256, 0, stream>>>(dw3, w3p);
    int total = NB * 256 * 256;
    dconv3_kernel<<<total / 256, 256, 0, stream>>>(d2, w3p, db3, out, total);
  }
}
